// Net_9998683865246
// MI455X (gfx1250) — matrix operand order repaired and checked by reading, not compiled, not run on hardware
//
#include <hip/hip_runtime.h>

// ---------------------------------------------------------------------------
// SNN forward (3-layer LIF) for MI455X / gfx1250, wave32 + WMMA f16.
//
// Phase 0: W1 (128x700 f32) -> W1h (128x704 f16, zero-padded) once in d_ws.
// Phase 1: cur1[b*T+t, 0:128] = x[b,t,:] @ W1^T   (one huge parallel GEMM,
//          x converted f32->f16 inline, W1 read pre-converted f16,
//          v_wmma_f32_16x16x32_f16, output f16 in d_ws).  Guard-free main
//          K loop; only the x-side tail chunk is value-masked.
// Phase 2: persistent scan kernel, 16 WGs x 16 batch rows; W2/W3 in LDS f16,
//          LIF membrane state in registers, per-step WMMA GEMMs for L2/L3.
// ---------------------------------------------------------------------------

typedef __attribute__((ext_vector_type(16))) _Float16 v16h;
typedef __attribute__((ext_vector_type(8)))  _Float16 v8h;
typedef __attribute__((ext_vector_type(8)))  float    v8f;

constexpr int kT    = 500;   // timesteps
constexpr int kB    = 256;   // batch
constexpr int kC    = 700;   // input features
constexpr int kCP   = 704;   // input features padded to 32
constexpr int kH1   = 128;   // layer-1 width
constexpr int kH2   = 256;   // layer-2 width
constexpr int kH3   = 20;    // layer-3 width
constexpr int kH3P  = 32;    // layer-3 width padded to WMMA N multiple

__device__ inline v8f wmma_f16(v16h a, v16h b, v8f c) {
  // D = A(16x32 f16) * B(32x16 f16) + C(16x16 f32)
  return __builtin_amdgcn_wmma_f32_16x16x32_f16(
      /*neg_a=*/false, a, /*neg_b=*/false, b,
      /*c_mod=*/(short)0, c, /*reuse_a=*/false, /*reuse_b=*/false);
}

// Unguarded: convert 4 consecutive f32 at p into d[e..e+3].
__device__ inline void cvt4u(v16h& d, int e, const float* p) {
  const float4 v = *(const float4*)p;
  d[e + 0] = (_Float16)v.x; d[e + 1] = (_Float16)v.y;
  d[e + 2] = (_Float16)v.z; d[e + 3] = (_Float16)v.w;
}

// Masked tail: address clamped (load always executes, no EXEC divergence),
// values zeroed when the chunk is out of range.  kC % 4 == 0, kk % 4 == 0,
// so each float4 chunk is entirely valid or entirely invalid.
__device__ inline void cvt4m(v16h& d, int e, const float* base, int kk) {
  const bool ok = kk < kC;
  const float4 v = *(const float4*)(base + (ok ? kk : 0));
  d[e + 0] = ok ? (_Float16)v.x : (_Float16)0.f;
  d[e + 1] = ok ? (_Float16)v.y : (_Float16)0.f;
  d[e + 2] = ok ? (_Float16)v.z : (_Float16)0.f;
  d[e + 3] = ok ? (_Float16)v.w : (_Float16)0.f;
}

// A-fragment (16x32 f16) gather from a row-major [16 x rowStride] f16 tile.
// Lane layout: lanes 0-15 row=lane, K = 0..15;
// lanes 16-31 row=lane-16, K = 16..31.  Both runs are 16B-aligned.
__device__ inline v16h load_a16x32(const _Float16* base, int rowStride,
                                   int k0, int row, int khalf) {
  const _Float16* p = base + row * rowStride + k0 + 16 * khalf;
  v8h lo = *(const v8h*)(p);
  v8h hi = *(const v8h*)(p + 8);
  v16h a;
#pragma unroll
  for (int e = 0; e < 8; ++e) { a[e] = lo[e]; a[8 + e] = hi[e]; }
  return a;
}

// B-fragment (32x16 f16): lane holds column n, 16 consecutive K values
// (lanes 0-15: K=k0..k0+15, lanes 16-31: K=k0+16..k0+31).  wrow points at the
// start of f16 weight row n (W[n][k] contiguous == B[k][n] column).
__device__ inline v16h load_b32x16(const _Float16* wrow, int koff) {
  v8h lo = *(const v8h*)(wrow + koff);
  v8h hi = *(const v8h*)(wrow + koff + 8);
  v16h b;
#pragma unroll
  for (int e = 0; e < 8; ++e) { b[e] = lo[e]; b[8 + e] = hi[e]; }
  return b;
}

// ---------------------------------------------------------------------------
// Phase 0: one-time W1 f32 -> f16 with zero padding to K=704.
// ---------------------------------------------------------------------------
__global__ void __launch_bounds__(256) snn_cvtw1_kernel(
    const float* __restrict__ W1,      // [128, 700]
    _Float16* __restrict__ W1h) {      // [128, 704]
  const int i = blockIdx.x * 256 + threadIdx.x;
  if (i < kH1 * kCP) {
    const int n = i / kCP;
    const int k = i - n * kCP;
    W1h[i] = (k < kC) ? (_Float16)W1[n * kC + k] : (_Float16)0.f;
  }
}

// ---------------------------------------------------------------------------
// Phase 1: cur1 = x @ W1^T   ([128000 x 700] @ [700 x 128], K padded to 704)
// Grid: 8000 WGs x 128 threads (4 waves).  Each WG -> 16x128 output tile,
// wave w -> N columns [32w, 32w+32).
// ---------------------------------------------------------------------------
__global__ void __launch_bounds__(128) snn_gemm1_kernel(
    const float* __restrict__ x,        // [B*T, 700] f32
    const _Float16* __restrict__ W1h,   // [128, 704] f16 (pre-converted)
    _Float16* __restrict__ cur1) {      // [B*T, 128] f16
  const int lane  = threadIdx.x & 31;
  const int wave  = threadIdx.x >> 5;
  const int row   = lane & 15;
  const int khalf = lane >> 4;
  const long r0   = (long)blockIdx.x * 16;
  const int  n0   = wave * 32;

  const float*    xrow  = x   + (r0 + row) * (long)kC;
  const _Float16* w0row = W1h + (long)(n0 + row) * kCP;
  const _Float16* w1row = W1h + (long)(n0 + 16 + row) * kCP;

  v8f acc0 = {}; v8f acc1 = {};

  // ---- main K loop: k0 = 0..640, max touched index 671 < 700: no guards ----
  for (int kt = 0; kt < 21; ++kt) {
    const int k0 = kt * 32;
    __builtin_prefetch(xrow + k0 + 128, 0, 1);   // stream-ahead on x

    const int ka = k0 + 16 * khalf;  // A: 16 consecutive K ka..ka+15
    v16h a;
    cvt4u(a, 0,  xrow + ka);
    cvt4u(a, 4,  xrow + ka + 4);
    cvt4u(a, 8,  xrow + ka + 8);
    cvt4u(a, 12, xrow + ka + 12);

    const int kb = k0 + 16 * khalf;  // B: 16 consecutive K, f16 direct
    const v16h bA = load_b32x16(w0row, kb);
    const v16h bB = load_b32x16(w1row, kb);

    acc0 = wmma_f16(a, bA, acc0);
    acc1 = wmma_f16(a, bB, acc1);
  }

  // ---- tail chunk: k0 = 672, covers 672..703; mask x beyond 700 ----
  {
    const int k0 = 672;
    const int ka = k0 + 16 * khalf;
    v16h a;
    cvt4m(a, 0,  xrow, ka);
    cvt4m(a, 4,  xrow, ka + 4);
    cvt4m(a, 8,  xrow, ka + 8);
    cvt4m(a, 12, xrow, ka + 12);

    const int kb = k0 + 16 * khalf;            // W1h is zero-padded: no mask
    const v16h bA = load_b32x16(w0row, kb);
    const v16h bB = load_b32x16(w1row, kb);

    acc0 = wmma_f16(a, bA, acc0);
    acc1 = wmma_f16(a, bB, acc1);
  }

  // D layout: VGPR g, lanes 0-15 -> M=g, lanes 16-31 -> M=g+8; N = lane&15.
#pragma unroll
  for (int g = 0; g < 8; ++g) {
    const long rr = r0 + g + 8 * khalf;
    cur1[rr * kH1 + n0 + row]      = (_Float16)acc0[g];
    cur1[rr * kH1 + n0 + 16 + row] = (_Float16)acc1[g];
  }
}

// ---------------------------------------------------------------------------
// Phase 2: persistent LIF scan.  Grid: 16 WGs x 256 threads (8 waves).
// WG owns batch rows [16*wg, 16*wg+16) for all 500 steps (independent in B).
// ---------------------------------------------------------------------------
__global__ void __launch_bounds__(256) snn_scan_kernel(
    const _Float16* __restrict__ cur1,   // [B*T, 128] f16 (phase-1 output)
    const float* __restrict__ b1, const float* __restrict__ W2,
    const float* __restrict__ b2, const float* __restrict__ W3,
    const float* __restrict__ b3,
    const float* __restrict__ beta1p, const float* __restrict__ beta2p,
    const float* __restrict__ beta3p,
    const float* __restrict__ thr1p, const float* __restrict__ thr2p,
    const float* __restrict__ thr3p,
    float* __restrict__ out) {           // [B, 20]
  extern __shared__ __align__(16) char smem[];
  _Float16* W2h = (_Float16*)smem;            // [256][128] f16
  _Float16* W3h = W2h + kH2 * kH1;            // [32][256]  f16 (rows >=20 zero)
  _Float16* s1h = W3h + kH3P * kH2;           // [16][128]  f16 spikes L1
  _Float16* s2h = s1h + 16 * kH1;             // [16][256]  f16 spikes L2
  float* cur2s  = (float*)(s2h + 16 * kH2);   // [16][256]  f32 GEMM2 out
  float* cur3s  = cur2s + 16 * kH2;           // [16][32]   f32 GEMM3 out

  const int tid   = threadIdx.x;
  const int lane  = tid & 31;
  const int wave  = tid >> 5;
  const int col   = lane & 15;
  const int khalf = lane >> 4;
  const int b0    = blockIdx.x * 16;

  // Stage weights to LDS as f16.
  for (int i = tid; i < kH2 * kH1; i += 256) W2h[i] = (_Float16)W2[i];
  for (int i = tid; i < kH3P * kH2; i += 256)
    W3h[i] = (i < kH3 * kH2) ? (_Float16)W3[i] : (_Float16)0.f;

  const float bt1 = fminf(fmaxf(beta1p[0], 0.f), 1.f);
  const float bt2 = fminf(fmaxf(beta2p[0], 0.f), 1.f);
  const float bt3 = fminf(fmaxf(beta3p[0], 0.f), 1.f);
  const float th1 = thr1p[0], th2 = thr2p[0], th3 = thr3p[0];

  // Per-thread persistent membrane state (fixed element ownership across t).
  float m1r[8], m2r[16], m3r[2], a3r[2];
#pragma unroll
  for (int j = 0; j < 8;  ++j) m1r[j] = 0.f;
#pragma unroll
  for (int j = 0; j < 16; ++j) m2r[j] = 0.f;
#pragma unroll
  for (int j = 0; j < 2;  ++j) { m3r[j] = 0.f; a3r[j] = 0.f; }

  const int   n1  = tid & (kH1 - 1);                 // LIF1 column
  const float b1c = b1[n1];
  const float b2c = b2[tid];                         // LIF2 column == tid
  const int   n3  = tid & (kH3P - 1);                // LIF3 column
  const float b3c = (n3 < kH3) ? b3[n3] : 0.f;

  __syncthreads();

  for (int t = 0; t < kT; ++t) {
    // ---- LIF layer 1 (elementwise; reads cur1 from global) ----
#pragma unroll
    for (int j = 0; j < 8; ++j) {
      const int e = tid + j * 256;
      const int i = e >> 7;                          // tile row 0..15
      const float cur =
          (float)cur1[((long)(b0 + i) * kT + t) * kH1 + n1] + b1c;
      const float mo  = m1r[j];
      const float rst = (mo - th1 > 0.f) ? th1 : 0.f;  // reset from OLD mem
      const float mn  = bt1 * mo + cur - rst;
      m1r[j] = mn;
      s1h[i * kH1 + n1] = (_Float16)((mn - th1 > 0.f) ? 1.f : 0.f);
    }
    __syncthreads();

    // ---- GEMM2: [16x128] @ [128x256]; wave w -> N in [32w, 32w+32) ----
    {
      const int nn0 = wave * 32;
      v8f acc0 = {}; v8f acc1 = {};
#pragma unroll
      for (int kt2 = 0; kt2 < kH1 / 32; ++kt2) {
        const int k0 = kt2 * 32;
        const v16h a   = load_a16x32(s1h, kH1, k0, col, khalf);
        const v16h bb0 = load_b32x16(W2h + (nn0 + col) * kH1,
                                     k0 + 16 * khalf);
        acc0 = wmma_f16(a, bb0, acc0);
        const v16h bb1 = load_b32x16(W2h + (nn0 + 16 + col) * kH1,
                                     k0 + 16 * khalf);
        acc1 = wmma_f16(a, bb1, acc1);
      }
#pragma unroll
      for (int g = 0; g < 8; ++g) {
        const int m = g + 8 * khalf;
        cur2s[m * kH2 + nn0 + col]      = acc0[g];
        cur2s[m * kH2 + nn0 + 16 + col] = acc1[g];
      }
    }
    __syncthreads();

    // ---- LIF layer 2 ----
#pragma unroll
    for (int j = 0; j < 16; ++j) {
      const float cur = cur2s[j * kH2 + tid] + b2c;
      const float mo  = m2r[j];
      const float rst = (mo - th2 > 0.f) ? th2 : 0.f;
      const float mn  = bt2 * mo + cur - rst;
      m2r[j] = mn;
      s2h[j * kH2 + tid] = (_Float16)((mn - th2 > 0.f) ? 1.f : 0.f);
    }
    __syncthreads();

    // ---- GEMM3: [16x256] @ [256x32]; waves 0-1 only (uniform branch) ----
    if (wave < 2) {
      const int nn0 = wave * 16;
      v8f acc = {};
#pragma unroll
      for (int kt3 = 0; kt3 < kH2 / 32; ++kt3) {
        const int k0 = kt3 * 32;
        const v16h a  = load_a16x32(s2h, kH2, k0, col, khalf);
        const v16h bb = load_b32x16(W3h + (nn0 + col) * kH2,
                                    k0 + 16 * khalf);
        acc = wmma_f16(a, bb, acc);
      }
#pragma unroll
      for (int g = 0; g < 8; ++g)
        cur3s[(g + 8 * khalf) * kH3P + nn0 + col] = acc[g];
    }
    __syncthreads();

    // ---- LIF layer 3 + spike accumulation ----
#pragma unroll
    for (int j = 0; j < 2; ++j) {
      const int e = tid + j * 256;
      const float cur = cur3s[e] + b3c;
      const float mo  = m3r[j];
      const float rst = (mo - th3 > 0.f) ? th3 : 0.f;
      const float mn  = bt3 * mo + cur - rst;
      m3r[j] = mn;
      a3r[j] += (mn - th3 > 0.f) ? 1.f : 0.f;
    }
    __syncthreads();
  }

  // ---- write mean over T ----
#pragma unroll
  for (int j = 0; j < 2; ++j) {
    const int e = tid + j * 256;
    const int i = e >> 5;                     // tile row 0..15
    if (n3 < kH3)
      out[(b0 + i) * kH3 + n3] = a3r[j] * (1.f / (float)kT);
  }
}

// ---------------------------------------------------------------------------
extern "C" void kernel_launch(void* const* d_in, const int* in_sizes, int n_in,
                              void* d_out, int out_size, void* d_ws,
                              size_t ws_size, hipStream_t stream) {
  (void)in_sizes; (void)n_in; (void)out_size; (void)ws_size;

  const float* x     = (const float*)d_in[0];
  const float* W1    = (const float*)d_in[1];
  const float* b1    = (const float*)d_in[2];
  const float* W2    = (const float*)d_in[3];
  const float* b2    = (const float*)d_in[4];
  const float* W3    = (const float*)d_in[5];
  const float* b3    = (const float*)d_in[6];
  const float* beta1 = (const float*)d_in[7];
  const float* beta2 = (const float*)d_in[8];
  const float* beta3 = (const float*)d_in[9];
  const float* thr1  = (const float*)d_in[10];
  const float* thr2  = (const float*)d_in[11];
  const float* thr3  = (const float*)d_in[12];
  float* out = (float*)d_out;

  // Workspace layout (f16):
  //   cur1 : [B*T, 128]  = 32,768,000 B (16B aligned)
  //   W1h  : [128, 704]  =    180,224 B
  _Float16* cur1 = (_Float16*)d_ws;
  _Float16* W1h  = cur1 + (size_t)kB * kT * kH1;

  // Phase 0: one-time W1 -> f16 (zero-padded K).
  snn_cvtw1_kernel<<<dim3((kH1 * kCP + 255) / 256), dim3(256), 0, stream>>>(
      W1, W1h);

  // Phase 1: 128000 rows / 16 per WG = 8000 WGs, 128 threads (4 waves).
  snn_gemm1_kernel<<<dim3((kB * kT) / 16), dim3(128), 0, stream>>>(
      x, W1h, cur1);

  // Phase 2: persistent scan, 16 WGs x 256 threads, ~112.6 KB dynamic LDS.
  const size_t lds_bytes =
      (size_t)(kH2 * kH1 + kH3P * kH2 + 16 * kH1 + 16 * kH2) * sizeof(_Float16)
      + (size_t)(16 * kH2 + 16 * kH3P) * sizeof(float);
  hipFuncSetAttribute(reinterpret_cast<const void*>(snn_scan_kernel),
                      hipFuncAttributeMaxDynamicSharedMemorySize,
                      (int)lds_bytes);
  snn_scan_kernel<<<dim3(kB / 16), dim3(256), lds_bytes, stream>>>(
      cur1, b1, W2, b2, W3, b3, beta1, beta2, beta3, thr1, thr2, thr3, out);
}
